// SimpleRNN_37709812859258
// MI455X (gfx1250) — compile-verified, matrix operand order checked by reading, not run on hardware
//
#include <hip/hip_runtime.h>
#include <hip/hip_bf16.h>
#include <stdint.h>

// Problem dims (fixed by the reference)
#define T_STEPS 8192
#define D_IN    2048
#define H_DIM   2048
#define G3      (3 * H_DIM)   // 6144

#define SCAN_BLOCKS 128

// GEMM block tiling
#define MBLK 64
#define NBLK 128
#define KCH  32
#define NTILES_N (G3 / NBLK)      // 48
#define KSTEPS   (D_IN / KCH)     // 64

typedef __attribute__((ext_vector_type(16))) __bf16 v16bf;
typedef __attribute__((ext_vector_type(8)))  __bf16 v8bf;
typedef __attribute__((ext_vector_type(8)))  float  v8f;

__device__ __forceinline__ float bf2f(unsigned short u) {
    return __uint_as_float(((unsigned int)u) << 16);
}
__device__ __forceinline__ unsigned short f2bf(float f) {
    unsigned int x = __float_as_uint(f);
    unsigned int r = (x + 0x7FFFu + ((x >> 16) & 1u)) >> 16;  // RNE
    return (unsigned short)r;
}

// Raw 32-bit LDS byte offset of a __shared__ object (what the async
// instruction's LDS-address VGPR wants).
__device__ __forceinline__ unsigned lds_off(const void* p) {
    return (unsigned)(unsigned long long)
        (__attribute__((address_space(3))) const void*)p;
}

// gfx1250 async global->LDS copy, 16B per lane. Tracked by ASYNCcnt.
__device__ __forceinline__ void async_copy_b128(unsigned lds_byte_off,
                                                const void* gaddr) {
    asm volatile("global_load_async_to_lds_b128 %0, %1, off"
                 :: "v"(lds_byte_off), "v"(gaddr) : "memory");
}
__device__ __forceinline__ void wait_async_le(int) {}  // (doc helper, unused)
__device__ __forceinline__ void s_wait_async_3() {
    asm volatile("s_wait_asynccnt 0x3" ::: "memory");
}
__device__ __forceinline__ void s_wait_async_0() {
    asm volatile("s_wait_asynccnt 0x0" ::: "memory");
}

// ---------------- Phase 0: f32 -> bf16 conversion ----------------
__global__ void cvt_f32_bf16(const float* __restrict__ in,
                             unsigned short* __restrict__ out, int n) {
    int i = blockIdx.x * blockDim.x + threadIdx.x;
    if (i < n) out[i] = f2bf(in[i]);
}

// ---------------- Phase 1: igates = xs @ Wih^T + bias ----------------
// 256 threads = 8 waves arranged 4(M) x 2(N); wave computes 16x64; block 64x128.
// Double-buffered LDS tiles filled via async global->LDS b128 copies.
__global__ __launch_bounds__(256)
void gru_igates_gemm(const unsigned short* __restrict__ xsb,   // [T, D] bf16
                     const unsigned short* __restrict__ wihb,  // [3H, D] bf16
                     const float* __restrict__ bias,           // [3H]
                     float* __restrict__ igates)               // [T, 3H]
{
    __shared__ __align__(16) unsigned short As[2][MBLK][KCH];   // 2 x 4 KB
    __shared__ __align__(16) unsigned short Bs[2][NBLK][KCH];   // 2 x 8 KB

    const int tid  = threadIdx.x;
    const int lane = tid & 31;
    const int warp = tid >> 5;

    const int mt = blockIdx.x / NTILES_N;   // 0..127
    const int nt = blockIdx.x % NTILES_N;   // 0..47
    const int m0 = mt * MBLK;
    const int n0 = nt * NBLK;

    const int wm = warp & 3;    // wave row 0..3  -> 16 M rows each
    const int wn = warp >> 2;   // wave col 0..1  -> 64 N cols each
    const int lm   = lane & 15;
    const int hsel = lane >> 4; // k-half select per CDNA5 WMMA layout

    // ---- async copy source/dest mapping (exact coverage per chunk) ----
    // A tile: 256 chunks of 16B -> 1 per thread. row = tid>>2, part = tid&3.
    const int arow = tid >> 2, apart = tid & 3;
    const unsigned short* agp = xsb + (size_t)(m0 + arow) * D_IN + apart * 8;
    // B tile: 512 chunks -> 2 per thread (rows r and r+64).
    const int brow = tid >> 2, bpart = tid & 3;
    const unsigned short* bgp0 = wihb + (size_t)(n0 + brow)      * D_IN + bpart * 8;
    const unsigned short* bgp1 = wihb + (size_t)(n0 + brow + 64) * D_IN + bpart * 8;

    unsigned aoff[2], b0off[2], b1off[2];
#pragma unroll
    for (int b = 0; b < 2; ++b) {
        aoff[b]  = lds_off(&As[b][arow][apart * 8]);
        b0off[b] = lds_off(&Bs[b][brow][bpart * 8]);
        b1off[b] = lds_off(&Bs[b][brow + 64][bpart * 8]);
    }

    // Accumulators pre-loaded with bias[n] (bias add is free)
    v8f acc[4];
#pragma unroll
    for (int j = 0; j < 4; ++j) {
        float bv = bias[n0 + wn * 64 + j * 16 + lm];
#pragma unroll
        for (int r = 0; r < 8; ++r) acc[j][r] = bv;
    }

    // Prologue: fill buffer 0 for k-chunk 0
    async_copy_b128(aoff[0],  agp);
    async_copy_b128(b0off[0], bgp0);
    async_copy_b128(b1off[0], bgp1);

    for (int kc = 0; kc < KSTEPS; ++kc) {
        const int buf = kc & 1;
        const int kb_next = (kc + 1) * KCH;
        if (kb_next < D_IN) {
            // Issue next chunk into the other buffer, then wait for CURRENT
            // chunk only (<=3 leaves the next chunk's 3 copies in flight).
            async_copy_b128(aoff[buf ^ 1],  agp  + kb_next);
            async_copy_b128(b0off[buf ^ 1], bgp0 + kb_next);
            async_copy_b128(b1off[buf ^ 1], bgp1 + kb_next);
            s_wait_async_3();
        } else {
            s_wait_async_0();
        }
        __syncthreads();   // all waves' copies for `buf` complete

        // A fragment 16x32 from LDS: lane<16 K={0..7,16..23}, lane>=16 +8.
        const unsigned short* ar = &As[buf][wm * 16 + lm][0];
        v8bf alo = *reinterpret_cast<const v8bf*>(ar + hsel * 8);
        v8bf ahi = *reinterpret_cast<const v8bf*>(ar + 16 + hsel * 8);
        v16bf a = __builtin_shufflevector(alo, ahi,
                      0, 1, 2, 3, 4, 5, 6, 7, 8, 9, 10, 11, 12, 13, 14, 15);
#pragma unroll
        for (int j = 0; j < 4; ++j) {
            const unsigned short* br = &Bs[buf][wn * 64 + j * 16 + lm][0];
            v8bf blo = *reinterpret_cast<const v8bf*>(br + hsel * 8);
            v8bf bhi = *reinterpret_cast<const v8bf*>(br + 16 + hsel * 8);
            v16bf b = __builtin_shufflevector(blo, bhi,
                          0, 1, 2, 3, 4, 5, 6, 7, 8, 9, 10, 11, 12, 13, 14, 15);
            acc[j] = __builtin_amdgcn_wmma_f32_16x16x32_bf16(
                         false, a, false, b, (short)0, acc[j], false, false);
        }
        __syncthreads();   // done reading `buf` before it is refilled
    }

    // C/D layout: lane L -> n = L&15 ; rows (L>>4)*8 + r in vgpr r.
    const int mrow0 = m0 + wm * 16 + hsel * 8;
#pragma unroll
    for (int j = 0; j < 4; ++j) {
        int n = n0 + wn * 64 + j * 16 + lm;
#pragma unroll
        for (int r = 0; r < 8; ++r)
            igates[(size_t)(mrow0 + r) * G3 + n] = acc[j][r];
    }
}

// ---------------- Phase 2 init: h0 = 0, barrier counter = 0 ----------------
__global__ void scan_init(float* hbuf, unsigned int* counter) {
    int i = blockIdx.x * blockDim.x + threadIdx.x;
    if (i < 2 * H_DIM) hbuf[i] = 0.f;
    if (i == 0) *counter = 0u;
}

// ---------------- Phase 2: persistent sequential GRU scan ----------------
// 128 blocks x 256 threads. Block b owns h indices [16b, 16b+16).
// Whh (bf16, 25 MB) stays L2-resident; h ping-pongs in global, staged in LDS.
__global__ __launch_bounds__(256)
void gru_scan(const unsigned short* __restrict__ whhb,  // [3H, H] bf16
              const float* __restrict__ igates,         // [T, 3H]
              const float* __restrict__ bias_n,         // [H]
              float* hbuf,                               // [2, H] ping-pong
              float* __restrict__ out,                   // [H]
              unsigned int* counter)
{
    __shared__ float lh[H_DIM];
    const int tid  = threadIdx.x;
    const int lane = tid & 31;
    const int warp = tid >> 5;
    const int blk  = blockIdx.x;
    const int j0   = blk * 16 + warp * 2;

    const int rows[6] = { j0,     j0 + H_DIM,     j0 + 2 * H_DIM,
                          j0 + 1, j0 + 1 + H_DIM, j0 + 1 + 2 * H_DIM };

    for (int t = 0; t < T_STEPS; ++t) {
        volatile const float* hcur = hbuf + (t & 1) * H_DIM;
        float* hnext = hbuf + ((t + 1) & 1) * H_DIM;

        for (int i = tid; i < H_DIM; i += 256) lh[i] = hcur[i];
        __syncthreads();

        float red[6];
#pragma unroll
        for (int r = 0; r < 6; ++r) {
            const ushort2* wr = reinterpret_cast<const ushort2*>(
                whhb + (size_t)rows[r] * H_DIM);
            float s = 0.f;
            for (int i = lane; i < H_DIM / 2; i += 32) {
                ushort2 wv = wr[i];
                s = fmaf(bf2f(wv.x), lh[2 * i],     s);
                s = fmaf(bf2f(wv.y), lh[2 * i + 1], s);
            }
#pragma unroll
            for (int off = 16; off > 0; off >>= 1)
                s += __shfl_xor(s, off, 32);
            red[r] = s;
        }

        if (lane < 2) {
            int j = j0 + lane;
            float hr = (lane == 0) ? red[0] : red[3];
            float hz = (lane == 0) ? red[1] : red[4];
            float hn = (lane == 0) ? red[2] : red[5];
            const float* ig = igates + (size_t)t * G3;
            float ir  = ig[j];
            float iz  = ig[j + H_DIM];
            float inn = ig[j + 2 * H_DIM];
            float rg = 1.f / (1.f + __expf(-(ir + hr)));
            float zg = 1.f / (1.f + __expf(-(iz + hz)));
            float ng = tanhf(inn + rg * (hn + bias_n[j]));
            float hv = (1.f - zg) * ng + zg * lh[j];
            hnext[j] = hv;
        }

        __syncthreads();
        if (tid == 0) {
            __threadfence();
            atomicAdd(counter, 1u);
            unsigned int target = (unsigned int)(t + 1) * SCAN_BLOCKS;
            while (__hip_atomic_load(counter, __ATOMIC_ACQUIRE,
                                     __HIP_MEMORY_SCOPE_AGENT) < target) { }
            __threadfence();
        }
        __syncthreads();
    }

    const float* hf = hbuf + (T_STEPS & 1) * H_DIM;
    if (tid < 16) out[blk * 16 + tid] = hf[blk * 16 + tid];
}

// ---------------- Host launch ----------------
extern "C" void kernel_launch(void* const* d_in, const int* in_sizes, int n_in,
                              void* d_out, int out_size, void* d_ws, size_t ws_size,
                              hipStream_t stream) {
    const float* xs     = (const float*)d_in[0];  // [T, D]
    const float* Wih    = (const float*)d_in[1];  // [3H, D]
    const float* Whh    = (const float*)d_in[2];  // [3H, H]
    const float* bias   = (const float*)d_in[3];  // [3H]
    const float* bias_n = (const float*)d_in[4];  // [H]
    float* out = (float*)d_out;

    char* ws = (char*)d_ws;
    size_t off = 0;
    auto carve = [&](size_t bytes) -> void* {
        void* p = ws + off;
        off = (off + bytes + 255) & ~(size_t)255;
        return p;
    };

    unsigned short* xsb    = (unsigned short*)carve((size_t)T_STEPS * D_IN * 2);
    unsigned short* wihb   = (unsigned short*)carve((size_t)G3 * D_IN * 2);
    unsigned short* whhb   = (unsigned short*)carve((size_t)G3 * H_DIM * 2);
    float*          igates = (float*)carve((size_t)T_STEPS * G3 * 4);
    float*          hbuf   = (float*)carve((size_t)2 * H_DIM * 4);
    unsigned int*   ctr    = (unsigned int*)carve(256);
    (void)ws_size; (void)in_sizes; (void)n_in; (void)out_size;

    {
        int n = T_STEPS * D_IN;
        cvt_f32_bf16<<<(n + 255) / 256, 256, 0, stream>>>(xs, xsb, n);
    }
    {
        int n = G3 * D_IN;
        cvt_f32_bf16<<<(n + 255) / 256, 256, 0, stream>>>(Wih, wihb, n);
    }
    {
        int n = G3 * H_DIM;
        cvt_f32_bf16<<<(n + 255) / 256, 256, 0, stream>>>(Whh, whhb, n);
    }

    // 128 M-tiles x 48 N-tiles = 6144 blocks
    gru_igates_gemm<<<6144, 256, 0, stream>>>(xsb, wihb, bias, igates);

    scan_init<<<(2 * H_DIM + 255) / 256, 256, 0, stream>>>(hbuf, ctr);
    gru_scan<<<SCAN_BLOCKS, 256, 0, stream>>>(whhb, igates, bias_n, hbuf, out, ctr);
}
